// Regressor_15281493639425
// MI455X (gfx1250) — matrix operand order repaired and checked by reading, not compiled, not run on hardware
//
#include <hip/hip_runtime.h>

typedef __attribute__((ext_vector_type(16))) _Float16 v16h;
typedef __attribute__((ext_vector_type(8)))  _Float16 v8h;
typedef __attribute__((ext_vector_type(8)))  float    v8f;

// GCC-style int4 for the async-LDS builtin signature (vector_size, not ext_vector)
typedef int v4i_vs __attribute__((vector_size(16)));
typedef v4i_vs __attribute__((address_space(1))) gv4i;   // global (AS1)
typedef v4i_vs __attribute__((address_space(3))) lv4i;   // LDS (AS3)

#define HH   448
#define WW   512
#define NCLS 128
#define HW   (HH*WW)      // 229376
#define LDX  136          // padded LDS row (halves): 272B = 17*16B, keeps 16B alignment

__device__ __forceinline__ float lrelu(float v) { return v >= 0.f ? v : 0.01f * v; }

// K pair-base for VGPR slot i of a 16-bit A fragment (16x32), per CDNA5 ISA layout
__device__ __forceinline__ int kbA(int i, int half) {
    return (i < 4 ? 2 * i : 16 + 2 * (i - 4)) + 8 * half;
}
// K pair-base for VGPR slot i of a 16-bit B fragment (32x16): same k order as A
__device__ __forceinline__ int kbB(int i, int half) {
    return (i < 4 ? 2 * i : 16 + 2 * (i - 4)) + 8 * half;
}

// ---------------------------------------------------------------------------
// Stage a packed-f16 column-major tile [256 cols][64 k] into LDS.
// Source rows are 256B contiguous -> use CDNA5 async global->LDS B128 copies
// (ASYNCcnt-tracked) when the toolchain exposes them; else 16B vector copies.
// ---------------------------------------------------------------------------
__device__ __forceinline__ void stage_f16_tile(_Float16 (*Xs)[LDX],
                                               const _Float16* __restrict__ src,
                                               int tid)
{
#if __has_builtin(__builtin_amdgcn_global_load_async_to_lds_b128)
    for (int c = tid; c < 2048; c += 256) {             // 2048 x 16B = 32KB tile
        int n = c >> 3, ko = (c & 7) * 8;
        __builtin_amdgcn_global_load_async_to_lds_b128(
            (gv4i*)(src + (size_t)n * 128 + ko),
            (lv4i*)&Xs[n][ko],
            0, 0);
    }
#if __has_builtin(__builtin_amdgcn_s_wait_asynccnt)
    __builtin_amdgcn_s_wait_asynccnt(0);
#else
    asm volatile("s_wait_asynccnt 0x0" ::: "memory");
#endif
#else
    for (int c = tid; c < 2048; c += 256) {
        int n = c >> 3, ko = (c & 7) * 8;
        *(v8h*)&Xs[n][ko] = *(const v8h*)(src + (size_t)n * 128 + ko);
    }
#endif
}

// ---------------------------------------------------------------------------
// One WMMA pass: this wave's 16 output rows x 128 cols =
//   W(16x128 stripe) @ X(128x128, staged in LDS column-major).
// Epilogue: +bias, lrelu, store COLUMN-MAJOR [col][row] as 16B packs.
// ---------------------------------------------------------------------------
__device__ __forceinline__ void gemm_pass(const _Float16 (*Xs)[LDX],
                                          const float* __restrict__ wgt_h,   // 128x128 rows of W[h]
                                          const float* __restrict__ bias_h,  // 128 biases of h
                                          _Float16* __restrict__ outT_h,     // [512][128] of h
                                          int n0, int m0, int lr, int half)
{
    v8f acc[8] = {};
    const float* wrow = wgt_h + (size_t)(m0 + lr) * 128;

    #pragma unroll
    for (int ks = 0; ks < 4; ++ks) {
        const int kofs = ks * 32;
        v16h a;
        #pragma unroll
        for (int i = 0; i < 8; ++i) {
            int kb = kofs + kbA(i, half);
            a[2 * i]     = (_Float16)wrow[kb];
            a[2 * i + 1] = (_Float16)wrow[kb + 1];
        }
        #pragma unroll
        for (int j = 0; j < 8; ++j) {
            const _Float16* xcol = &Xs[j * 16 + lr][0];
            v16h b;
            #pragma unroll
            for (int i = 0; i < 8; ++i) {
                int kb = kofs + kbB(i, half);
                b[2 * i]     = xcol[kb];
                b[2 * i + 1] = xcol[kb + 1];
            }
            acc[j] = __builtin_amdgcn_wmma_f32_16x16x32_f16(
                false, a, false, b, (short)0, acc[j], false, false);
        }
    }

    const int row0 = m0 + 8 * half;
    #pragma unroll
    for (int j = 0; j < 8; ++j) {
        int col = n0 + j * 16 + lr;
        v8h pack;
        #pragma unroll
        for (int v = 0; v < 8; ++v)
            pack[v] = (_Float16)lrelu(acc[j][v] + bias_h[row0 + v]);
        *(v8h*)&outT_h[(size_t)col * 128 + row0] = pack;   // contiguous 16B store
    }
}

// ---------------------------------------------------------------------------
// Fused first layer: stage x_in tile once (f32 -> f16, transposed), then run
// TWO weight sets against it: t1 = lrelu(Wcl1@X+b), reg = lrelu(Wreg1@X+b).
// Saves a full 117MB re-read of x_in. A barrier between the passes keeps the
// scheduler from interleaving them (was pushing the wave past 256 VGPRs).
// ---------------------------------------------------------------------------
__global__ __launch_bounds__(256)
void fused_x_kernel(const float* __restrict__ xin,
                    const float* __restrict__ w1, const float* __restrict__ b1,
                    const float* __restrict__ wr, const float* __restrict__ br,
                    _Float16* __restrict__ t1T, _Float16* __restrict__ regT)
{
    __shared__ _Float16 Xs[128][LDX];

    const int h   = blockIdx.y;
    const int n0  = blockIdx.x * 128;
    const int tid = threadIdx.x;

    for (int e = tid; e < 128 * 128; e += 256) {          // transpose + cvt staging
        int k = e >> 7, n = e & 127;
        Xs[n][k] = (_Float16)xin[(size_t)k * HW + (size_t)h * WW + n0 + n];
    }
    __syncthreads();

    const int wave = tid >> 5, lane = tid & 31;
    const int lr = lane & 15, half = lane >> 4, m0 = wave * 16;

    gemm_pass(Xs, w1 + (size_t)h * 128 * 128, b1 + (size_t)h * 128,
              t1T + (size_t)h * WW * 128, n0, m0, lr, half);

    __syncthreads();   // scheduling fence: retire pass-1 regs before pass 2

    gemm_pass(Xs, wr + (size_t)h * 128 * 128, br + (size_t)h * 128,
              regT + (size_t)h * WW * 128, n0, m0, lr, half);
}

// ---------------------------------------------------------------------------
// Middle layer: t2 = lrelu(Wcl2 @ t1 + b). Input already column-major f16.
// ---------------------------------------------------------------------------
__global__ __launch_bounds__(256)
void gemm_mid_kernel(const _Float16* __restrict__ inT,
                     const float* __restrict__ wgt, const float* __restrict__ bias,
                     _Float16* __restrict__ outT)
{
    __shared__ _Float16 Xs[128][LDX];

    const int h   = blockIdx.y;
    const int n0  = blockIdx.x * 128;
    const int tid = threadIdx.x;

    stage_f16_tile(Xs, inT + ((size_t)h * WW + n0) * 128, tid);
    __syncthreads();

    const int wave = tid >> 5, lane = tid & 31;
    const int lr = lane & 15, half = lane >> 4, m0 = wave * 16;

    gemm_pass(Xs, wgt + (size_t)h * 128 * 128, bias + (size_t)h * 128,
              outT + (size_t)h * WW * 128, n0, m0, lr, half);
}

// ---------------------------------------------------------------------------
// Classifier GEMM (129 rows) with fused argmax + mask; `cl` never hits global.
// ---------------------------------------------------------------------------
__global__ __launch_bounds__(256)
void gemm_cls_kernel(const _Float16* __restrict__ inT, const float* __restrict__ wgt,
                     const float* __restrict__ bias, int* __restrict__ inds,
                     float* __restrict__ mask_out)
{
    __shared__ _Float16 Xs[128][LDX];
    __shared__ float candv[8][128];
    __shared__ int   candi[8][128];

    const int h   = blockIdx.y;
    const int n0  = blockIdx.x * 128;
    const int tid = threadIdx.x;

    stage_f16_tile(Xs, inT + ((size_t)h * WW + n0) * 128, tid);
    __syncthreads();

    const int wave = tid >> 5, lane = tid & 31;
    const int lr = lane & 15, half = lane >> 4, m0 = wave * 16;

    v8f acc[8] = {};
    const float* wrow = wgt + ((size_t)h * 129 + (m0 + lr)) * 128;

    #pragma unroll
    for (int ks = 0; ks < 4; ++ks) {
        const int kofs = ks * 32;
        v16h a;
        #pragma unroll
        for (int i = 0; i < 8; ++i) {
            int kb = kofs + kbA(i, half);
            a[2 * i]     = (_Float16)wrow[kb];
            a[2 * i + 1] = (_Float16)wrow[kb + 1];
        }
        #pragma unroll
        for (int j = 0; j < 8; ++j) {
            const _Float16* xcol = &Xs[j * 16 + lr][0];
            v16h b;
            #pragma unroll
            for (int i = 0; i < 8; ++i) {
                int kb = kofs + kbB(i, half);
                b[2 * i]     = xcol[kb];
                b[2 * i + 1] = xcol[kb + 1];
            }
            acc[j] = __builtin_amdgcn_wmma_f32_16x16x32_f16(
                false, a, false, b, (short)0, acc[j], false, false);
        }
    }

    // per-wave argmax over this wave's 16 rows, per column
    #pragma unroll
    for (int j = 0; j < 8; ++j) {
        float best = -3.4e38f; int bi = 0;
        #pragma unroll
        for (int v = 0; v < 8; ++v) {
            int row = m0 + v + 8 * half;
            float val = acc[j][v] + bias[h * 129 + row];
            if (val > best) { best = val; bi = row; }
        }
        float ov = __shfl_xor(best, 16, 32);
        int   oi = __shfl_xor(bi,   16, 32);
        if (ov > best || (ov == best && oi < bi)) { best = ov; bi = oi; }
        if (half == 0) { candv[wave][j * 16 + lr] = best; candi[wave][j * 16 + lr] = bi; }
    }
    __syncthreads();

    if (tid < 128) {
        const int n = tid;
        float best = candv[0][n]; int bi = candi[0][n];
        #pragma unroll
        for (int wv = 1; wv < 8; ++wv) {
            float v = candv[wv][n];
            if (v > best) { best = v; bi = candi[wv][n]; }   // strict > keeps lowest index
        }
        inds[h * WW + n0 + n] = bi;

        // mask row (129th output row): scalar dot over LDS tile (contiguous reads)
        const float* w128 = wgt + ((size_t)h * 129 + 128) * 128;
        float s = bias[h * 129 + 128];
        for (int k = 0; k < 128; ++k) s += (float)Xs[n][k] * w128[k];
        mask_out[h * WW + n0 + n] = lrelu(s);
    }
}

// ---------------------------------------------------------------------------
// Per-pixel regressor head. One wave32 per flat row m = w*448 + h.
// Features are column-major -> 128 contiguous f16 per pixel (coalesced).
// Reproduces the reference's mixed indexing: features from (h=m%448, w=m/448),
// gather index from h-major flat m (inds[m] + (m/512)*128).
// ---------------------------------------------------------------------------
__global__ __launch_bounds__(256)
void head_kernel(const _Float16* __restrict__ regT, const int* __restrict__ inds,
                 const float* __restrict__ w2, const float* __restrict__ b2,
                 const float* __restrict__ w3, const float* __restrict__ b3,
                 float* __restrict__ out)
{
    const int wid  = (int)((blockIdx.x * (unsigned)blockDim.x + threadIdx.x) >> 5);
    const int lane = threadIdx.x & 31;
    if (wid >= HW) return;

    const int m    = wid;
    const int h    = m % HH;
    const int wcol = m / HH;
    const int isel = inds[m] + (m / WW) * NCLS;

    const _Float16* feat = regT + ((size_t)h * WW + wcol) * 128;
    const float*    w2r  = w2 + (size_t)isel * (128 * 4);
    float s0 = 0.f, s1 = 0.f, s2 = 0.f, s3 = 0.f;
    #pragma unroll
    for (int t = 0; t < 4; ++t) {
        int c = lane + t * 32;
        float f = (float)feat[c];
        const float* wr = w2r + c * 4;
        s0 += f * wr[0]; s1 += f * wr[1]; s2 += f * wr[2]; s3 += f * wr[3];
    }
    #pragma unroll
    for (int off = 16; off >= 1; off >>= 1) {
        s0 += __shfl_xor(s0, off, 32);
        s1 += __shfl_xor(s1, off, 32);
        s2 += __shfl_xor(s2, off, 32);
        s3 += __shfl_xor(s3, off, 32);
    }
    if (lane == 0) {
        const float* b2r = b2 + (size_t)isel * 4;
        float h0 = lrelu(s0 + b2r[0]);
        float h1 = lrelu(s1 + b2r[1]);
        float h2 = lrelu(s2 + b2r[2]);
        float h3 = lrelu(s3 + b2r[3]);
        const float* w3r = w3 + (size_t)isel * 4;
        float r = h0 * w3r[0] + h1 * w3r[1] + h2 * w3r[2] + h3 * w3r[3] + b3[isel];
        const int pix = h * WW + wcol;
        out[pix] = ((float)inds[pix] + r) * (1.0f / (float)NCLS);
    }
}

// ---------------------------------------------------------------------------
extern "C" void kernel_launch(void* const* d_in, const int* in_sizes, int n_in,
                              void* d_out, int out_size, void* d_ws, size_t ws_size,
                              hipStream_t stream)
{
    (void)in_sizes; (void)n_in; (void)out_size; (void)ws_size;

    const float* x_in   = (const float*)d_in[0];
    const float* w_cl1  = (const float*)d_in[1];
    const float* b_cl1  = (const float*)d_in[2];
    const float* w_cl2  = (const float*)d_in[3];
    const float* b_cl2  = (const float*)d_in[4];
    const float* w_cl3  = (const float*)d_in[5];
    const float* b_cl3  = (const float*)d_in[6];
    const float* w_reg1 = (const float*)d_in[7];
    const float* b_reg1 = (const float*)d_in[8];
    const float* w2     = (const float*)d_in[9];
    const float* b2     = (const float*)d_in[10];
    const float* w3     = (const float*)d_in[11];
    const float* b3     = (const float*)d_in[12];
    float* out = (float*)d_out;

    const size_t FBUF = (size_t)HH * 128 * WW * sizeof(_Float16);   // 58,720,256 B
    char* ws = (char*)d_ws;
    _Float16* t1T  = (_Float16*)(ws);
    _Float16* t2T  = (_Float16*)(ws + FBUF);
    _Float16* regT = (_Float16*)(ws + 2 * FBUF);
    int*      inds = (int*)    (ws + 3 * FBUF);

    dim3 grid(WW / 128, HH), blk(256);

    fused_x_kernel<<<grid, blk, 0, stream>>>(x_in, w_cl1, b_cl1, w_reg1, b_reg1, t1T, regT);
    gemm_mid_kernel<<<grid, blk, 0, stream>>>(t1T, w_cl2, b_cl2, t2T);
    gemm_cls_kernel<<<grid, blk, 0, stream>>>(t2T, w_cl3, b_cl3, inds, out + HW);

    const int total_threads = HW * 32;                 // one wave per pixel
    head_kernel<<<(total_threads + 255) / 256, 256, 0, stream>>>(
        regT, inds, w2, b2, w3, b3, out);
}
